// DDIEncoder_69423851373253
// MI455X (gfx1250) — compile-verified, matrix operand order checked by reading, not run on hardware
//
#include <hip/hip_runtime.h>
#include <stdint.h>

#define DEV __device__ __forceinline__

static constexpr int N_ = 10000, E_ = 50000;

typedef __attribute__((ext_vector_type(16))) __bf16 bf16x16;
typedef __attribute__((ext_vector_type(8)))  float  f32x8;

union FragB { uint4 q[2]; bf16x16 v; };

DEV uint32_t pkmul_bf16(uint32_t a, uint32_t b) {
  uint32_t d;
  asm("v_pk_mul_bf16 %0, %1, %2" : "=v"(d) : "v"(a), "v"(b));
  return d;
}

DEV unsigned short f2bf(float f) {  // round-to-nearest-even f32 -> bf16
  union { float f; uint32_t u; } c; c.f = f;
  uint32_t r = (c.u + 0x7fffu + ((c.u >> 16) & 1u)) >> 16;
  return (unsigned short)r;
}

// A = h-scaled x row: 8 packed bf16 muls build a 16x32 A fragment for this lane
DEV bf16x16 scale_frag(uint32_t hp, const uint4& p0, const uint4& p1) {
  FragB u;
  u.q[0].x = pkmul_bf16(hp, p0.x);
  u.q[0].y = pkmul_bf16(hp, p0.y);
  u.q[0].z = pkmul_bf16(hp, p0.z);
  u.q[0].w = pkmul_bf16(hp, p0.w);
  u.q[1].x = pkmul_bf16(hp, p1.x);
  u.q[1].y = pkmul_bf16(hp, p1.y);
  u.q[1].z = pkmul_bf16(hp, p1.z);
  u.q[1].w = pkmul_bf16(hp, p1.w);
  return u.v;
}

// ---------------- small utility kernels ----------------

__global__ void k_zero(float* p, int n) {
  int i = blockIdx.x * 256 + threadIdx.x;
  if (i < n) p[i] = 0.0f;
}

__global__ void k_cnt(const int* __restrict__ dst, float* __restrict__ cnt, int e_n) {
  int e = blockIdx.x * 256 + threadIdx.x;
  if (e < e_n) atomicAdd(&cnt[dst[e]], 1.0f);
}

// f32 [rows,cin] -> bf16 [rows,cpad] (zero-padded columns)
__global__ void k_cvt_pad(const float* __restrict__ src, unsigned short* __restrict__ out,
                          int rows, int cin, int cpad) {
  int i = blockIdx.x * 256 + threadIdx.x;
  if (i >= rows * cpad) return;
  int r = i / cpad, c = i - r * cpad;
  out[i] = f2bf(c < cin ? src[r * cin + c] : 0.0f);
}

// w [K,64] row-major f32 -> out [64,Kpad] column-major bf16 (K-pairs contiguous for b128 loads)
__global__ void k_cvt_t(const float* __restrict__ wsrc, unsigned short* __restrict__ out,
                        int K, int Kpad) {
  int i = blockIdx.x * 256 + threadIdx.x;
  if (i >= 64 * Kpad) return;
  int c = i / Kpad, k = i - c * Kpad;
  out[i] = f2bf(k < K ? wsrc[k * 64 + c] : 0.0f);
}

// per-column sum & sum-of-squares over [rows,64]
__global__ void k_colstats(const float* __restrict__ X, int rows, int rpb,
                           float* __restrict__ sum, float* __restrict__ sumsq) {
  __shared__ float ls[256], lq[256];
  const int col = threadIdx.x & 63;
  const int sub = threadIdx.x >> 6;  // 0..3
  int r = blockIdx.x * rpb + sub;
  int rend = (blockIdx.x + 1) * rpb; if (rend > rows) rend = rows;
  float s = 0.f, q = 0.f;
  for (; r < rend; r += 4) { float v = X[r * 64 + col]; s += v; q += v * v; }
  ls[threadIdx.x] = s; lq[threadIdx.x] = q;
  __syncthreads();
  if (sub == 0) {
    s = ls[col] + ls[col + 64] + ls[col + 128] + ls[col + 192];
    q = lq[col] + lq[col + 64] + lq[col + 128] + lq[col + 192];
    atomicAdd(&sum[col], s);
    atomicAdd(&sumsq[col], q);
  }
}

__global__ void k_bnparams(const float* __restrict__ sum, const float* __restrict__ sumsq,
                           const float* __restrict__ g, const float* __restrict__ b,
                           float inv_rows, float* __restrict__ scale, float* __restrict__ shift) {
  int c = threadIdx.x;
  float m = sum[c] * inv_rows;
  float v = sumsq[c] * inv_rows - m * m;
  float sc = g[c] * rsqrtf(v + 1e-5f);
  scale[c] = sc;
  shift[c] = b[c] - m * sc;
}

// relu(batchnorm(X)) -> bf16 (and optional f32 copy)
__global__ void k_bnrelu(const float* __restrict__ X, const float* __restrict__ scale,
                         const float* __restrict__ shift, unsigned short* __restrict__ obf,
                         float* __restrict__ of32, int total) {
  int i = blockIdx.x * 256 + threadIdx.x;
  if (i >= total) return;
  int c = i & 63;
  float v = fmaxf(fmaf(X[i], scale[c], shift[c]), 0.0f);
  obf[i] = f2bf(v);
  if (of32) of32[i] = v;
}

// out[n,o] = bias[o] + sum_i Xin[n,i]*root[i,o] + scatter-mean(aggr)
template <int IN>
__global__ void k_node_out(const float* __restrict__ Xin, const float* __restrict__ root,
                           const float* __restrict__ aggr, const float* __restrict__ cnt,
                           const float* __restrict__ bias, float* __restrict__ out, int nN) {
  int i = blockIdx.x * 256 + threadIdx.x;
  if (i >= nN * 64) return;
  int n = i >> 6, o = i & 63;
  float s = bias[o];
  const float* xr = Xin + n * IN;
#pragma unroll
  for (int k = 0; k < IN; ++k) s = fmaf(xr[k], root[k * 64 + o], s);
  float cv = cnt[n];
  s += (cv > 0.0f) ? (aggr[i] / cv) : 0.0f;
  out[i] = s;
}

// ---------------- WMMA kernels ----------------

// Hidden[e,c] = edge_attr_bf16[e,0:96] @ W1t + b1[c]   (K=96 = 3 steps of 32)
__global__ void k_edge_mlp_wmma(const uint32_t* __restrict__ Abf,  // [E,48] u32 (bf16 pairs)
                                const uint32_t* __restrict__ Bt,   // [64,48] u32, col-major bf16
                                const float* __restrict__ b1,
                                float* __restrict__ Hout,          // [E,64] f32
                                int ntiles) {
  const int wave = blockIdx.x * (blockDim.x >> 5) + (threadIdx.x >> 5);
  if (wave >= ntiles) return;                    // wave-uniform: EXEC all-ones for WMMA
  const int lane = threadIdx.x & 31;
  const int lm = lane & 15, lh = lane >> 4;
  const int e = wave * 16 + lm;
  f32x8 acc[4];
#pragma unroll
  for (int nt = 0; nt < 4; ++nt) acc[nt] = (f32x8)0.0f;
  const uint32_t* arow = Abf + e * 48 + lh * 4;
#pragma unroll
  for (int ks = 0; ks < 3; ++ks) {
    FragB a;
    a.q[0] = *(const uint4*)(arow + ks * 16);
    a.q[1] = *(const uint4*)(arow + ks * 16 + 8);
#pragma unroll
    for (int nt = 0; nt < 4; ++nt) {
      const uint32_t* bcol = Bt + (nt * 16 + lm) * 48 + lh * 4 + ks * 16;
      FragB b;
      b.q[0] = *(const uint4*)(bcol);
      b.q[1] = *(const uint4*)(bcol + 8);
      acc[nt] = __builtin_amdgcn_wmma_f32_16x16x32_bf16(false, a.v, false, b.v,
                                                        (short)0, acc[nt], false, false);
    }
  }
#pragma unroll
  for (int nt = 0; nt < 4; ++nt) {
    const int c = nt * 16 + lm;
    const float bias = b1[c];
#pragma unroll
    for (int r = 0; r < 8; ++r)
      Hout[(wave * 16 + r + 8 * lh) * 64 + c] = acc[nt][r] + bias;
  }
}

// Fused outer-product GEMM + scatter:
//   msg[e,:] = (h_e ⊗ x_src[e]) @ w2_view[K,64] + x_src[e] @ b2_view[IN,64]
//   aggr[dst[e],:] += msg[e,:]   (atomic)
template <int IN>
__global__ void k_msg_wmma(const int* __restrict__ src, const int* __restrict__ dst,
                           const unsigned short* __restrict__ Hrelu,  // [E,64] bf16
                           const uint32_t* __restrict__ Xbf,          // [Nrows, IN/2] u32 bf16
                           const uint32_t* __restrict__ Wrt,          // [64, K/2] u32, col-major
                           const uint32_t* __restrict__ Brt,          // [64, IN/2] u32, col-major
                           float* __restrict__ aggr,                  // [N,64] f32
                           int ntiles) {
  constexpr int NH = IN / 32;
  constexpr int K = 64 * IN;
  const int wave = blockIdx.x * (blockDim.x >> 5) + (threadIdx.x >> 5);
  if (wave >= ntiles) return;
  const int lane = threadIdx.x & 31;
  const int lm = lane & 15, lh = lane >> 4;
  const int e = wave * 16 + lm;

  // cache this lane's slice of the gathered x row (bf16 pairs, A-fragment order)
  const int srow = src[e];
  const uint32_t* xbase = Xbf + srow * (IN / 2) + lh * 4;
  uint4 xc[NH][2];
#pragma unroll
  for (int h = 0; h < NH; ++h) {
    xc[h][0] = *(const uint4*)(xbase + h * 16);
    xc[h][1] = *(const uint4*)(xbase + h * 16 + 8);
  }

  const unsigned short* hrow = Hrelu + e * 64;
  const uint32_t* colb[4];
#pragma unroll
  for (int nt = 0; nt < 4; ++nt) {
    colb[nt] = Wrt + (nt * 16 + lm) * (K / 2) + lh * 4;
    __builtin_prefetch(colb[nt], 0, 1);   // global_prefetch_b8: warm L2 path for B columns
  }

  f32x8 acc[4];
#pragma unroll
  for (int nt = 0; nt < 4; ++nt) acc[nt] = (f32x8)0.0f;

  for (int khi = 0; khi < 64; ++khi) {
    const uint32_t hv = (uint32_t)hrow[khi];
    const uint32_t hp = hv | (hv << 16);     // broadcast h to bf16x2
#pragma unroll
    for (int hs = 0; hs < NH; ++hs) {
      const int ks = khi * NH + hs;          // 32-wide K step index
      FragB a; a.v = scale_frag(hp, xc[hs][0], xc[hs][1]);
#pragma unroll
      for (int nt = 0; nt < 4; ++nt) {
        FragB b;
        b.q[0] = *(const uint4*)(colb[nt] + ks * 16);
        b.q[1] = *(const uint4*)(colb[nt] + ks * 16 + 8);
        acc[nt] = __builtin_amdgcn_wmma_f32_16x16x32_bf16(false, a.v, false, b.v,
                                                          (short)0, acc[nt], false, false);
      }
    }
  }

  // fold in b2 term: extra K steps with A = x row, B = reshaped b2
#pragma unroll
  for (int hs = 0; hs < NH; ++hs) {
    FragB a; a.q[0] = xc[hs][0]; a.q[1] = xc[hs][1];
#pragma unroll
    for (int nt = 0; nt < 4; ++nt) {
      const uint32_t* bb = Brt + (nt * 16 + lm) * (IN / 2) + hs * 16 + lh * 4;
      FragB b;
      b.q[0] = *(const uint4*)(bb);
      b.q[1] = *(const uint4*)(bb + 8);
      acc[nt] = __builtin_amdgcn_wmma_f32_16x16x32_bf16(false, a.v, false, b.v,
                                                        (short)0, acc[nt], false, false);
    }
  }

  // scatter-add into destination nodes (C layout: vgpr r -> M=r+8*lh, N=lm)
#pragma unroll
  for (int r = 0; r < 8; ++r) {
    const int row = wave * 16 + r + 8 * lh;
    const int dn = dst[row];
#pragma unroll
    for (int nt = 0; nt < 4; ++nt)
      atomicAdd(&aggr[dn * 64 + nt * 16 + lm], acc[nt][r]);
  }
}

// ---------------- launch ----------------

extern "C" void kernel_launch(void* const* d_in, const int* in_sizes, int n_in,
                              void* d_out, int out_size, void* d_ws, size_t ws_size,
                              hipStream_t stream) {
  (void)in_sizes; (void)n_in; (void)out_size; (void)ws_size;
  const float* x    = (const float*)d_in[0];
  const int*   ei   = (const int*)  d_in[1];
  const float* ea   = (const float*)d_in[2];
  const float* n1w1 = (const float*)d_in[3];
  const float* n1b1 = (const float*)d_in[4];
  const float* n1bg = (const float*)d_in[5];
  const float* n1bb = (const float*)d_in[6];
  const float* n1w2 = (const float*)d_in[7];
  const float* n1b2 = (const float*)d_in[8];
  const float* root1= (const float*)d_in[9];
  const float* bias1= (const float*)d_in[10];
  const float* bng  = (const float*)d_in[11];
  const float* bnb  = (const float*)d_in[12];
  const float* n2w1 = (const float*)d_in[13];
  const float* n2b1 = (const float*)d_in[14];
  const float* n2bg = (const float*)d_in[15];
  const float* n2bb = (const float*)d_in[16];
  const float* n2w2 = (const float*)d_in[17];
  const float* n2b2 = (const float*)d_in[18];
  const float* root2= (const float*)d_in[19];
  const float* bias2= (const float*)d_in[20];
  float* out = (float*)d_out;
  const int* src  = ei;
  const int* dstp = ei + E_;

  char* wp = (char*)d_ws;
  auto alloc = [&](size_t bytes) { void* p = (void*)wp; wp += (bytes + 255) & ~(size_t)255; return p; };
  unsigned short* Abf  = (unsigned short*)alloc((size_t)E_ * 96 * 2);
  unsigned short* Xbf  = (unsigned short*)alloc((size_t)N_ * 32 * 2);
  unsigned short* W1t1 = (unsigned short*)alloc(64 * 96 * 2);
  unsigned short* W1t2 = (unsigned short*)alloc(64 * 96 * 2);
  unsigned short* Wrt1 = (unsigned short*)alloc(64 * 2048 * 2);
  unsigned short* Wrt2 = (unsigned short*)alloc((size_t)64 * 4096 * 2);
  unsigned short* Brt1 = (unsigned short*)alloc(64 * 32 * 2);
  unsigned short* Brt2 = (unsigned short*)alloc(64 * 64 * 2);
  float* Hidden        = (float*)alloc((size_t)E_ * 64 * 4);
  unsigned short* Hrelu= (unsigned short*)alloc((size_t)E_ * 64 * 2);
  float* aggr          = (float*)alloc((size_t)N_ * 64 * 4);
  float* cnt           = (float*)alloc((size_t)N_ * 4);
  float* Out1          = (float*)alloc((size_t)N_ * 64 * 4);
  float* H1f           = (float*)alloc((size_t)N_ * 64 * 4);
  unsigned short* H1bf = (unsigned short*)alloc((size_t)N_ * 64 * 2);
  float* stats         = (float*)alloc(256 * 4);
  float* csum = stats, *csq = stats + 64, *scl = stats + 128, *shf = stats + 192;

  auto cdiv = [](int a, int b) { return (a + b - 1) / b; };
  const int TILES = E_ / 16;       // 3125 edge tiles
  const int GW = cdiv(TILES, 4);   // 4 waves / 128-thread block

  // one-time conversions / transposes (bf16, WMMA-friendly layouts)
  k_cvt_pad<<<cdiv(E_ * 96, 256), 256, 0, stream>>>(ea, Abf, E_, 86, 96);
  k_cvt_pad<<<cdiv(N_ * 32, 256), 256, 0, stream>>>(x, Xbf, N_, 32, 32);
  k_cvt_t<<<cdiv(64 * 96, 256), 256, 0, stream>>>(n1w1, W1t1, 86, 96);
  k_cvt_t<<<cdiv(64 * 96, 256), 256, 0, stream>>>(n2w1, W1t2, 86, 96);
  k_cvt_t<<<cdiv(64 * 2048, 256), 256, 0, stream>>>(n1w2, Wrt1, 2048, 2048);
  k_cvt_t<<<cdiv(64 * 4096, 256), 256, 0, stream>>>(n2w2, Wrt2, 4096, 4096);
  k_cvt_t<<<cdiv(64 * 32, 256), 256, 0, stream>>>(n1b2, Brt1, 32, 32);
  k_cvt_t<<<cdiv(64 * 64, 256), 256, 0, stream>>>(n2b2, Brt2, 64, 64);
  k_zero<<<cdiv(N_, 256), 256, 0, stream>>>(cnt, N_);
  k_cnt<<<cdiv(E_, 256), 256, 0, stream>>>(dstp, cnt, E_);

  // ---------------- layer 1 ----------------
  k_edge_mlp_wmma<<<GW, 128, 0, stream>>>((const uint32_t*)Abf, (const uint32_t*)W1t1, n1b1, Hidden, TILES);
  k_zero<<<1, 256, 0, stream>>>(stats, 128);
  k_colstats<<<cdiv(E_, 2048), 256, 0, stream>>>(Hidden, E_, 2048, csum, csq);
  k_bnparams<<<1, 64, 0, stream>>>(csum, csq, n1bg, n1bb, 1.0f / E_, scl, shf);
  k_bnrelu<<<cdiv(E_ * 64, 256), 256, 0, stream>>>(Hidden, scl, shf, Hrelu, (float*)nullptr, E_ * 64);
  k_zero<<<cdiv(N_ * 64, 256), 256, 0, stream>>>(aggr, N_ * 64);
  k_msg_wmma<32><<<GW, 128, 0, stream>>>(src, dstp, Hrelu, (const uint32_t*)Xbf,
                                         (const uint32_t*)Wrt1, (const uint32_t*)Brt1, aggr, TILES);
  k_node_out<32><<<cdiv(N_ * 64, 256), 256, 0, stream>>>(x, root1, aggr, cnt, bias1, Out1, N_);
  k_zero<<<1, 256, 0, stream>>>(stats, 128);
  k_colstats<<<cdiv(N_, 2048), 256, 0, stream>>>(Out1, N_, 2048, csum, csq);
  k_bnparams<<<1, 64, 0, stream>>>(csum, csq, bng, bnb, 1.0f / N_, scl, shf);
  k_bnrelu<<<cdiv(N_ * 64, 256), 256, 0, stream>>>(Out1, scl, shf, H1bf, H1f, N_ * 64);

  // ---------------- layer 2 ----------------
  k_edge_mlp_wmma<<<GW, 128, 0, stream>>>((const uint32_t*)Abf, (const uint32_t*)W1t2, n2b1, Hidden, TILES);
  k_zero<<<1, 256, 0, stream>>>(stats, 128);
  k_colstats<<<cdiv(E_, 2048), 256, 0, stream>>>(Hidden, E_, 2048, csum, csq);
  k_bnparams<<<1, 64, 0, stream>>>(csum, csq, n2bg, n2bb, 1.0f / E_, scl, shf);
  k_bnrelu<<<cdiv(E_ * 64, 256), 256, 0, stream>>>(Hidden, scl, shf, Hrelu, (float*)nullptr, E_ * 64);
  k_zero<<<cdiv(N_ * 64, 256), 256, 0, stream>>>(aggr, N_ * 64);
  k_msg_wmma<64><<<GW, 128, 0, stream>>>(src, dstp, Hrelu, (const uint32_t*)H1bf,
                                         (const uint32_t*)Wrt2, (const uint32_t*)Brt2, aggr, TILES);
  k_node_out<64><<<cdiv(N_ * 64, 256), 256, 0, stream>>>(H1f, root2, aggr, cnt, bias2, out, N_);
}
